// PointGenerator_33354716021243
// MI455X (gfx1250) — matrix operand order repaired and checked by reading, not compiled, not run on hardware
//
#include <hip/hip_runtime.h>
#include <hip/hip_bf16.h>

typedef _Float16 v16h __attribute__((ext_vector_type(16)));
typedef _Float16 v8h  __attribute__((ext_vector_type(8)));
typedef float    v8f  __attribute__((ext_vector_type(8)));

#define C_DIM 384
#define NC    2048
#define NT    8192

// ---------------- prep: tgt points = repeat(pred,4) + 0.02*noise ----------------
__global__ void prep_tgt_kernel(const float* __restrict__ pred,
                                const float* __restrict__ noise,
                                float* __restrict__ tgtP, int n) {
  int t = blockIdx.x * blockDim.x + threadIdx.x;
  if (t >= n) return;
  int r = t / 3, d = t - r * 3;
  tgtP[t] = pred[(r >> 2) * 3 + d] + 0.02f * noise[t];
}

// ---------------- f32 -> f16 convert ----------------
__global__ void f2h_kernel(const float* __restrict__ in, _Float16* __restrict__ out, int n) {
  int t = blockIdx.x * blockDim.x + threadIdx.x;
  if (t >= n) return;
  out[t] = (_Float16)in[t];
}

// ---------------- weight convert + transpose: in (Kd x N, row stride ld) -> out (N x Kd) f16 ----------------
__global__ void wconv_kernel(const float* __restrict__ in, int ld, int Kd, int N,
                             _Float16* __restrict__ out) {
  int t = blockIdx.x * blockDim.x + threadIdx.x;
  if (t >= Kd * N) return;
  int n = t / Kd, k = t - n * Kd;
  out[t] = (_Float16)in[(size_t)k * ld + n];
}

// ---------------- brute-force KNN (16-NN, self excluded) ----------------
// 64 queries per block, candidate range split 4 ways per query (tid = qlocal*4 + split).
// Per-thread top-16 lists in LDS (bd[s*256+tid]: consecutive lanes -> consecutive banks).
// Requires N multiple of 1024 (2048, 8192 ok).
__global__ __launch_bounds__(256) void knn_kernel(const float* __restrict__ P, int N,
                                                  int* __restrict__ oidx) {
  __shared__ float tx[256], ty[256], tz[256], ts[256];
  __shared__ float bd[16 * 256];
  __shared__ int   bj[16 * 256];
  int tid = threadIdx.x;
  int q   = blockIdx.x * 64 + (tid >> 2);
  int spl = tid & 3;
  float xi = P[q * 3], yi = P[q * 3 + 1], zi = P[q * 3 + 2];
  float sqi = xi * xi + yi * yi + zi * zi;
  const float INF = __builtin_inff();
#pragma unroll
  for (int s = 0; s < 16; ++s) { bd[s * 256 + tid] = INF; bj[s * 256 + tid] = 0; }
  float cmax = INF; int cslot = 0;
  int chunk = N >> 2;                 // candidates per split (tile-aligned)
  int j0 = spl * chunk, j1 = j0 + chunk;
  for (int t0 = 0; t0 < N; t0 += 256) {
    __syncthreads();
    { int j = t0 + tid;
      float x = P[j * 3], y = P[j * 3 + 1], z = P[j * 3 + 2];
      tx[tid] = x; ty[tid] = y; tz[tid] = z; ts[tid] = x * x + y * y + z * z; }
    __syncthreads();
    if (t0 >= j0 && t0 < j1) {
      for (int jj = 0; jj < 256; ++jj) {
        int j = t0 + jj;
        float d = sqi + ts[jj] - 2.0f * (xi * tx[jj] + yi * ty[jj] + zi * tz[jj]);
        if (j != q && d < cmax) {
          bd[cslot * 256 + tid] = d; bj[cslot * 256 + tid] = j;
          float m = -INF; int ms = 0;
#pragma unroll
          for (int s = 0; s < 16; ++s) { float v = bd[s * 256 + tid]; if (v > m) { m = v; ms = s; } }
          cmax = m; cslot = ms;
        }
      }
    }
  }
  __syncthreads();
  if (spl == 0) {
    // merge partial lists of splits 1..3 into split 0's list (ascending split order
    // keeps lowest-index winner on distance ties, matching stable top_k)
    for (int o = 1; o < 4; ++o) {
#pragma unroll
      for (int s = 0; s < 16; ++s) {
        float d = bd[s * 256 + tid + o];
        if (d < cmax) {
          bd[cslot * 256 + tid] = d; bj[cslot * 256 + tid] = bj[s * 256 + tid + o];
          float m = -INF; int ms = 0;
#pragma unroll
          for (int s2 = 0; s2 < 16; ++s2) { float v = bd[s2 * 256 + tid]; if (v > m) { m = v; ms = s2; } }
          cmax = m; cslot = ms;
        }
      }
    }
#pragma unroll
    for (int s = 0; s < 16; ++s) oidx[q * 16 + s] = bj[s * 256 + tid];
  }
}

// ---------------- edge epilogue: g[i] = mean_k relu(A[i>>s]-B[i>>s]+b1 + B[j>>s] + dxyz@W1c) ----------------
__global__ __launch_bounds__(384) void edge_kernel(const float* __restrict__ A,
                                                   const float* __restrict__ B,
                                                   const float* __restrict__ b1,
                                                   const float* __restrict__ W1c, // 3 x C, row stride C (f32)
                                                   const float* __restrict__ P,
                                                   const int* __restrict__ idx,
                                                   int shift, int C,
                                                   _Float16* __restrict__ g) {
  __shared__ int   sj[16];
  __shared__ float sdx[16], sdy[16], sdz[16];
  int i = blockIdx.x, c = threadIdx.x;
  if (c < 16) {
    int j = idx[i * 16 + c];
    sj[c]  = j;
    sdx[c] = P[j * 3]     - P[i * 3];
    sdy[c] = P[j * 3 + 1] - P[i * 3 + 1];
    sdz[c] = P[j * 3 + 2] - P[i * 3 + 2];
  }
  __syncthreads();
  int ia = i >> shift;
  float base = A[(size_t)ia * C + c] - B[(size_t)ia * C + c] + b1[c];
  float w0 = W1c[c], w1 = W1c[C + c], w2 = W1c[2 * C + c];
  float acc = 0.f;
#pragma unroll
  for (int k = 0; k < 16; ++k) {
    int j = sj[k] >> shift;
    float z = base + B[(size_t)j * C + c] + sdx[k] * w0 + sdy[k] * w1 + sdz[k] * w2;
    acc += fmaxf(z, 0.f);
  }
  g[(size_t)i * C + c] = (_Float16)(acc * 0.0625f);
}

// ---------------- WMMA GEMM: out = act(A(MxK) @ Wt^T + bias [+ xyz@Waux]) ----------------
// A row-major f16 (lda=Kd). Wt = W^T, f16 row-major (N x Kd).
// One wave computes a 16x64 strip: 4 accumulators share one A fragment per k-step
// (4x A reuse, 4 WMMAs per 10 b128 loads). Requires N % 64 == 0, Kd % 32 == 0.
__global__ __launch_bounds__(256) void wmma_gemm_kernel(
    const _Float16* __restrict__ A, const _Float16* __restrict__ Wt,
    const float* __restrict__ bias, const float* __restrict__ xyz,
    const float* __restrict__ Waux, _Float16* __restrict__ out16,
    float* __restrict__ out32, int M, int N, int Kd, int relu) {
  int lane = threadIdx.x & 31;
  int wave = threadIdx.x >> 5;
  int stripsN = N >> 6;                          // 64-wide strips
  int totalStrips = (M >> 4) * stripsN;
  int strip = blockIdx.x * (blockDim.x >> 5) + wave;
  if (strip >= totalStrips) return;              // uniform per wave: EXEC stays all-ones
  int tm = strip / stripsN, tn = strip - tm * stripsN;
  int l15 = lane & 15, hi = lane >> 4;

  // A frag: lane m, halves 0..15 -> K = 16*hi + 0..15
  const _Float16* ap = A + (size_t)(tm * 16 + l15) * Kd + 16 * hi;
  // B frags: lane n, halves 0..15 -> K = 16*hi + 0..15 (contiguous in W^T row)
  const _Float16* bp[4];
#pragma unroll
  for (int t = 0; t < 4; ++t)
    bp[t] = Wt + (size_t)(tn * 64 + t * 16 + l15) * Kd + 16 * hi;

  v8f acc[4];
#pragma unroll
  for (int t = 0; t < 4; ++t) acc[t] = (v8f){0.f, 0.f, 0.f, 0.f, 0.f, 0.f, 0.f, 0.f};

  for (int k = 0; k < Kd; k += 32) {
    v16h av;
    *(v8h*)&av       = *(const v8h*)(ap + k);
    *((v8h*)&av + 1) = *(const v8h*)(ap + k + 8);
#pragma unroll
    for (int t = 0; t < 4; ++t) {
      v16h bv = *(const v16h*)(bp[t] + k);
      acc[t] = __builtin_amdgcn_wmma_f32_16x16x32_f16(false, av, false, bv,
                                                      (short)0, acc[t], false, false);
    }
  }

#pragma unroll
  for (int t = 0; t < 4; ++t) {
    int col = tn * 64 + t * 16 + l15;
    float bi = bias ? bias[col] : 0.0f;
    float wa0 = 0.f, wa1 = 0.f, wa2 = 0.f;
    if (Waux) { wa0 = Waux[col]; wa1 = Waux[N + col]; wa2 = Waux[2 * N + col]; }
#pragma unroll
    for (int v = 0; v < 8; ++v) {
      int row = tm * 16 + v + 8 * hi;            // C/D layout: VGPR v -> M = v + 8*(lane>=16)
      float val = acc[t][v] + bi;
      if (Waux) val += xyz[row * 3] * wa0 + xyz[row * 3 + 1] * wa1 + xyz[row * 3 + 2] * wa2;
      if (relu) val = fmaxf(val, 0.f);
      if (out32) out32[(size_t)row * N + col] = val;
      if (out16) out16[(size_t)row * N + col] = (_Float16)val;
    }
  }
}

// ---------------- narrow head: out = base + scale*(act(MxKd) @ W(Kdx3) + b) ----------------
__global__ void dot3_kernel(const _Float16* __restrict__ act, const float* __restrict__ W,
                            const float* __restrict__ b, const float* __restrict__ base,
                            float scale, float* __restrict__ out, int M, int Kd) {
  int r = blockIdx.x * blockDim.x + threadIdx.x;
  if (r >= M) return;
  float a0 = 0.f, a1 = 0.f, a2 = 0.f;
  const _Float16* ar = act + (size_t)r * Kd;
  for (int k = 0; k < Kd; ++k) {
    float a = (float)ar[k];
    a0 += a * W[k * 3]; a1 += a * W[k * 3 + 1]; a2 += a * W[k * 3 + 2];
  }
  out[r * 3]     = base[r * 3]     + scale * (a0 + b[0]);
  out[r * 3 + 1] = base[r * 3 + 1] + scale * (a1 + b[1]);
  out[r * 3 + 2] = base[r * 3 + 2] + scale * (a2 + b[2]);
}

extern "C" void kernel_launch(void* const* d_in, const int* in_sizes, int n_in,
                              void* d_out, int out_size, void* d_ws, size_t ws_size,
                              hipStream_t stream) {
  const float* ctx_xyz = (const float*)d_in[0];
  const float* ctx_tok = (const float*)d_in[1];
  const float* pred    = (const float*)d_in[2];
  const float* noise   = (const float*)d_in[3];
  const float* W_se1 = (const float*)d_in[4];  const float* b_se1 = (const float*)d_in[5];
  const float* W_se2 = (const float*)d_in[6];  const float* b_se2 = (const float*)d_in[7];
  const float* W_cd1 = (const float*)d_in[8];  const float* b_cd1 = (const float*)d_in[9];
  const float* W_cd2 = (const float*)d_in[10]; const float* b_cd2 = (const float*)d_in[11];
  const float* W_f1  = (const float*)d_in[12]; const float* b_f1  = (const float*)d_in[13];
  const float* W_f2  = (const float*)d_in[14]; const float* b_f2  = (const float*)d_in[15];
  const float* W_f3  = (const float*)d_in[16]; const float* b_f3  = (const float*)d_in[17];
  const float* W_r1  = (const float*)d_in[18]; const float* b_r1  = (const float*)d_in[19];
  const float* W_r2  = (const float*)d_in[20]; const float* b_r2  = (const float*)d_in[21];
  float* out = (float*)d_out;

  char* ws = (char*)d_ws; size_t off = 0;
  auto alloc = [&](size_t bytes) -> void* {
    void* p = ws + off; off += (bytes + 255) & ~(size_t)255; return p;
  };
  float*     tgtP   = (float*)alloc((size_t)NT * 3 * 4);
  float*     tgtP2  = (float*)alloc((size_t)NT * 3 * 4);
  _Float16*  tok16  = (_Float16*)alloc((size_t)NC * C_DIM * 2);
  _Float16*  W1aT   = (_Float16*)alloc((size_t)C_DIM * C_DIM * 2);
  _Float16*  W1bT   = (_Float16*)alloc((size_t)C_DIM * C_DIM * 2);
  _Float16*  W2T    = (_Float16*)alloc((size_t)C_DIM * C_DIM * 2);
  _Float16*  Wcd1T  = (_Float16*)alloc((size_t)C_DIM * C_DIM * 2);
  _Float16*  Wf1bT  = (_Float16*)alloc((size_t)C_DIM * 256 * 2);
  _Float16*  Wf2T   = (_Float16*)alloc((size_t)256 * 256 * 2);
  _Float16*  Wr1aT  = (_Float16*)alloc((size_t)C_DIM * C_DIM * 2);
  _Float16*  Wr1bT  = (_Float16*)alloc((size_t)C_DIM * C_DIM * 2);
  int*       idx_c  = (int*)alloc((size_t)NC * 16 * 4);
  int*       idx_t  = (int*)alloc((size_t)NT * 16 * 4);
  int*       idx_r  = (int*)alloc((size_t)NT * 16 * 4);
  float*     Ac     = (float*)alloc((size_t)NC * C_DIM * 4);
  float*     Bc     = (float*)alloc((size_t)NC * C_DIM * 4);
  _Float16*  g_c16  = (_Float16*)alloc((size_t)NC * C_DIM * 2);
  _Float16*  latc16 = (_Float16*)alloc((size_t)NC * C_DIM * 2);
  _Float16*  hcd16  = (_Float16*)alloc((size_t)NC * C_DIM * 2);
  _Float16*  g_t16  = (_Float16*)alloc((size_t)NT * C_DIM * 2);
  _Float16*  latt16 = (_Float16*)alloc((size_t)NT * C_DIM * 2);
  _Float16*  h1_16  = (_Float16*)alloc((size_t)NT * 256 * 2);
  _Float16*  h2_16  = (_Float16*)alloc((size_t)NT * 256 * 2);
  float*     Ar     = (float*)alloc((size_t)NT * C_DIM * 4);
  float*     Br     = (float*)alloc((size_t)NT * C_DIM * 4);
  _Float16*  gr16   = (_Float16*)alloc((size_t)NT * C_DIM * 2);

  auto gemm = [&](const _Float16* A, const _Float16* Wt, const float* bias,
                  const float* xyz, const float* Waux, _Float16* o16, float* o32,
                  int M, int N, int Kd, int relu) {
    int strips = (M / 16) * (N / 64);
    wmma_gemm_kernel<<<(strips + 7) / 8, 256, 0, stream>>>(A, Wt, bias, xyz, Waux,
                                                           o16, o32, M, N, Kd, relu);
  };
  auto wconv = [&](const float* in, int ld, int Kd, int N, _Float16* o) {
    int n = Kd * N;
    wconv_kernel<<<(n + 255) / 256, 256, 0, stream>>>(in, ld, Kd, N, o);
  };

  // ---- setup ----
  prep_tgt_kernel<<<(NT * 3 + 255) / 256, 256, 0, stream>>>(pred, noise, tgtP, NT * 3);
  f2h_kernel<<<(NC * C_DIM + 255) / 256, 256, 0, stream>>>(ctx_tok, tok16, NC * C_DIM);
  wconv(W_se1,             384, 384, 384, W1aT);
  wconv(W_se1 + 384 * 384, 384, 384, 384, W1bT);
  wconv(W_se2,             384, 384, 384, W2T);
  wconv(W_cd1,             384, 384, 384, Wcd1T);
  wconv(W_f1 + 3 * 256,    256, 384, 256, Wf1bT);
  wconv(W_f2,              256, 256, 256, Wf2T);
  wconv(W_r1,              384, 384, 384, Wr1aT);
  wconv(W_r1 + 384 * 384,  384, 384, 384, Wr1bT);

  // ---- stage A: ctx edge_conv (W_se) ----
  knn_kernel<<<NC / 64, 256, 0, stream>>>(ctx_xyz, NC, idx_c);
  gemm(tok16, W1aT, nullptr, nullptr, nullptr, nullptr, Ac, NC, 384, 384, 0);
  gemm(tok16, W1bT, nullptr, nullptr, nullptr, nullptr, Bc, NC, 384, 384, 0);
  edge_kernel<<<NC, 384, 0, stream>>>(Ac, Bc, b_se1, W_se1 + 768 * 384, ctx_xyz,
                                      idx_c, 0, C_DIM, g_c16);
  gemm(g_c16, W2T, b_se2, nullptr, nullptr, latc16, nullptr, NC, 384, 384, 0);

  // ---- stage B: coord deform -> out rows [0, 2048) ----
  gemm(latc16, Wcd1T, b_cd1, nullptr, nullptr, hcd16, nullptr, NC, 384, 384, 1);
  dot3_kernel<<<(NC + 255) / 256, 256, 0, stream>>>(hcd16, W_cd2, b_cd2, ctx_xyz,
                                                    0.05f, out, NC, 384);

  // ---- stage C: tgt edge_conv (W_se, tokens gathered via r>>2) ----
  knn_kernel<<<NT / 64, 256, 0, stream>>>(tgtP, NT, idx_t);
  edge_kernel<<<NT, 384, 0, stream>>>(Ac, Bc, b_se1, W_se1 + 768 * 384, tgtP,
                                      idx_t, 2, C_DIM, g_t16);
  gemm(g_t16, W2T, b_se2, nullptr, nullptr, latt16, nullptr, NT, 384, 384, 0);

  // ---- stage D: fold MLP (rank-3 xyz term fused via Waux = W_f1 rows 0..2) ----
  gemm(latt16, Wf1bT, b_f1, tgtP, W_f1, h1_16, nullptr, NT, 256, 384, 1);
  gemm(h1_16, Wf2T, b_f2, nullptr, nullptr, h2_16, nullptr, NT, 256, 256, 1);
  dot3_kernel<<<(NT + 255) / 256, 256, 0, stream>>>(h2_16, W_f3, b_f3, tgtP,
                                                    1.0f, tgtP2, NT, 256);

  // ---- stage E: refine edge_conv (W_r) -> out rows [2048, 10240) ----
  knn_kernel<<<NT / 64, 256, 0, stream>>>(tgtP2, NT, idx_r);
  gemm(latt16, Wr1aT, nullptr, nullptr, nullptr, nullptr, Ar, NT, 384, 384, 0);
  gemm(latt16, Wr1bT, nullptr, nullptr, nullptr, nullptr, Br, NT, 384, 384, 0);
  edge_kernel<<<NT, 384, 0, stream>>>(Ar, Br, b_r1, W_r1 + 768 * 384, tgtP2,
                                      idx_r, 0, C_DIM, gr16);
  dot3_kernel<<<(NT + 255) / 256, 256, 0, stream>>>(gr16, W_r2, b_r2, tgtP2,
                                                    1.0f, out + (size_t)NC * 3, NT, 384);
}
